// H2GCN_8443905704363
// MI455X (gfx1250) — compile-verified, matrix operand order checked by reading, not run on hardware
//
#include <hip/hip_runtime.h>

#define BN_EPS 1e-5f

typedef __attribute__((ext_vector_type(16))) __bf16 v16bf;
typedef __attribute__((ext_vector_type(8)))  float  v8f;

union Frag16 {
    v16bf v;
    uint4 q[2];
};

__device__ __forceinline__ unsigned short f2bf(float f) {
    unsigned u = __float_as_uint(f);
    u += 0x7FFFu + ((u >> 16) & 1u);   // round-to-nearest-even
    return (unsigned short)(u >> 16);
}

// ---------------------------------------------------------------------------
// GEMM: Y[nrows x 128] = X[nrows x 128] @ W[128 x 128]   (bf16 WMMA, f32 acc)
// One block = 128 rows. Full W held in LDS (transposed, bf16).
// Per k-step: preload A + all 8 B fragments (ds_load_b128 clause), then run
// the 8 WMMAs back-to-back so DS latency overlaps matrix issue.
// ---------------------------------------------------------------------------
__global__ __launch_bounds__(256)
void gemm_nodes_bf16(const float* __restrict__ X, const float* __restrict__ W,
                     float* __restrict__ Y, int nrows)
{
    extern __shared__ unsigned short smem[];
    unsigned short* sA = smem;              // [128][136] row-major (rows x K)
    unsigned short* sB = smem + 128 * 136;  // [128][136] = W^T  (cols x K)

    const int tid  = threadIdx.x;
    const int row0 = blockIdx.x * 128;

    // Stage X tile (f32 -> bf16), zero-pad rows past nrows.
    for (int idx = tid; idx < 128 * 128; idx += 256) {
        int r = idx >> 7, k = idx & 127;
        int gr = row0 + r;
        float v = (gr < nrows) ? X[(size_t)gr * 128 + k] : 0.0f;
        sA[r * 136 + k] = f2bf(v);
    }
    // Stage W transposed (f32 -> bf16).
    for (int idx = tid; idx < 128 * 128; idx += 256) {
        int k = idx >> 7, n = idx & 127;
        sB[n * 136 + k] = f2bf(W[k * 128 + n]);
    }
    __syncthreads();

    const int wave = tid >> 5;
    const int lane = tid & 31;
    const int hf   = lane >> 4;   // lane half selects K sub-group
    const int l16  = lane & 15;
    const int m0   = wave * 16;   // 8 waves -> 8 M-subtiles of 16 rows

    v8f zacc = {};
    v8f acc[8];
#pragma unroll
    for (int nt = 0; nt < 8; ++nt) acc[nt] = zacc;

#pragma unroll
    for (int kt = 0; kt < 4; ++kt) {
        const int kb = kt * 32 + hf * 8;

        Frag16 a;
        const uint4* pa = (const uint4*)(sA + (m0 + l16) * 136 + kb);
        a.q[0] = pa[0];   // K = kb .. kb+7
        a.q[1] = pa[2];   // K = kb+16 .. kb+23

        Frag16 b[8];
#pragma unroll
        for (int nt = 0; nt < 8; ++nt) {
            const uint4* pb = (const uint4*)(sB + (nt * 16 + l16) * 136 + kb);
            b[nt].q[0] = pb[0];
            b[nt].q[1] = pb[2];
        }
#pragma unroll
        for (int nt = 0; nt < 8; ++nt) {
            acc[nt] = __builtin_amdgcn_wmma_f32_16x16x32_bf16(
                false, a.v, false, b[nt].v, (short)0, acc[nt], false, false);
        }
    }

#pragma unroll
    for (int nt = 0; nt < 8; ++nt) {
        int col = nt * 16 + l16;
#pragma unroll
        for (int v = 0; v < 8; ++v) {
            int row = row0 + m0 + v + 8 * hf;
            if (row < nrows) Y[(size_t)row * 128 + col] = acc[nt][v];
        }
    }
}

// ---------------------------------------------------------------------------
// FC: out[nrows x 40] = concat(X0,X1,X2)[nrows x 384] @ Wfc[384 x 40] + bfc
// N padded to 48 (3 tiles of 16); K = 384 = 12 WMMA k-steps.
// ---------------------------------------------------------------------------
__global__ __launch_bounds__(256)
void fc_concat_bf16(const float* __restrict__ X0, const float* __restrict__ X1,
                    const float* __restrict__ X2, const float* __restrict__ Wfc,
                    const float* __restrict__ bfc, float* __restrict__ Y, int nrows)
{
    extern __shared__ unsigned short smem[];
    unsigned short* sA = smem;              // [128][392]
    unsigned short* sB = smem + 128 * 392;  // [48][392] = Wfc^T, cols 40..47 zero

    const int tid  = threadIdx.x;
    const int row0 = blockIdx.x * 128;

    for (int idx = tid; idx < 128 * 384; idx += 256) {
        int r = idx / 384, c = idx - r * 384;
        int gr = row0 + r;
        float v = 0.0f;
        if (gr < nrows) {
            if (c < 128)      v = X0[(size_t)gr * 128 + c];
            else if (c < 256) v = X1[(size_t)gr * 128 + (c - 128)];
            else              v = X2[(size_t)gr * 128 + (c - 256)];
        }
        sA[r * 392 + c] = f2bf(v);
    }
    for (int idx = tid; idx < 48 * 384; idx += 256) {
        int k = idx / 48, n = idx - k * 48;
        float v = (n < 40) ? Wfc[k * 40 + n] : 0.0f;
        sB[n * 392 + k] = f2bf(v);
    }
    __syncthreads();

    const int wave = tid >> 5;
    const int lane = tid & 31;
    const int hf   = lane >> 4;
    const int l16  = lane & 15;
    const int m0   = wave * 16;

    v8f zacc = {};
    v8f acc[3];
#pragma unroll
    for (int nt = 0; nt < 3; ++nt) acc[nt] = zacc;

#pragma unroll
    for (int kt = 0; kt < 12; ++kt) {
        const int kb = kt * 32 + hf * 8;

        Frag16 a;
        const uint4* pa = (const uint4*)(sA + (m0 + l16) * 392 + kb);
        a.q[0] = pa[0];
        a.q[1] = pa[2];

        Frag16 b[3];
#pragma unroll
        for (int nt = 0; nt < 3; ++nt) {
            const uint4* pb = (const uint4*)(sB + (nt * 16 + l16) * 392 + kb);
            b[nt].q[0] = pb[0];
            b[nt].q[1] = pb[2];
        }
#pragma unroll
        for (int nt = 0; nt < 3; ++nt) {
            acc[nt] = __builtin_amdgcn_wmma_f32_16x16x32_bf16(
                false, a.v, false, b[nt].v, (short)0, acc[nt], false, false);
        }
    }

#pragma unroll
    for (int nt = 0; nt < 3; ++nt) {
        int col = nt * 16 + l16;
        if (col < 40) {
            float bias = bfc[col];
#pragma unroll
            for (int v = 0; v < 8; ++v) {
                int row = row0 + m0 + v + 8 * hf;
                if (row < nrows) Y[(size_t)row * 40 + col] = acc[nt][v] + bias;
            }
        }
    }
}

// ---------------------------------------------------------------------------
// Elementwise / scatter kernels
// ---------------------------------------------------------------------------
__global__ void zero_f32(float* p, int n)
{
    int i = blockIdx.x * blockDim.x + threadIdx.x;
    if (i < n) p[i] = 0.0f;
}

__global__ void deg_count(const int* __restrict__ row, float* __restrict__ deg, int E)
{
    int i = blockIdx.x * blockDim.x + threadIdx.x;
    if (i < E) atomicAdd(&deg[row[i]], 1.0f);
}

__global__ void deg_finish(float* deg, int n)
{
    int i = blockIdx.x * blockDim.x + threadIdx.x;
    if (i < n) deg[i] = rsqrtf(deg[i] + 1.0f);
}

// agg[i,f] = h[i,f] * dinv[i]^2 + bias[f]   (self-loop + bias init)
__global__ void init_agg(const float* __restrict__ h, const float* __restrict__ dinv,
                         const float* __restrict__ bias, float* __restrict__ agg, int n)
{
    int idx = blockIdx.x * blockDim.x + threadIdx.x;
    if (idx < n * 128) {
        int node = idx >> 7;
        int f = idx & 127;
        float di = dinv[node];
        agg[idx] = h[idx] * di * di + bias[f];
    }
}

// One wave (32 lanes) per edge: lane gathers float4 of h[col] (L2-resident,
// 51 MB << 192 MB L2), atomically accumulates norm-scaled message into agg[row].
__global__ __launch_bounds__(256)
void scatter_edges(const int* __restrict__ erow, const int* __restrict__ ecol,
                   const float* __restrict__ dinv, const float* __restrict__ h,
                   float* __restrict__ agg, int E)
{
    int e = blockIdx.x * 8 + (threadIdx.x >> 5);
    if (e >= E) return;
    int lane = threadIdx.x & 31;
    int r = erow[e];
    int c = ecol[e];
    float norm = dinv[r] * dinv[c];
    const float4* hs = (const float4*)(h + (size_t)c * 128);
    float4 v = hs[lane];
    float* dst = agg + (size_t)r * 128 + lane * 4;
    atomicAdd(dst + 0, v.x * norm);
    atomicAdd(dst + 1, v.y * norm);
    atomicAdd(dst + 2, v.z * norm);
    atomicAdd(dst + 3, v.w * norm);
}

__global__ void bn_relu(const float* __restrict__ agg,
                        const float* __restrict__ gamma, const float* __restrict__ beta,
                        const float* __restrict__ mean,  const float* __restrict__ var,
                        float* __restrict__ out, int n)
{
    int idx = blockIdx.x * blockDim.x + threadIdx.x;
    if (idx < n * 128) {
        int f = idx & 127;
        float s = gamma[f] * rsqrtf(var[f] + BN_EPS);
        float v = (agg[idx] - mean[f]) * s + beta[f];
        out[idx] = fmaxf(v, 0.0f);
    }
}

// ---------------------------------------------------------------------------
// Launcher
// ---------------------------------------------------------------------------
extern "C" void kernel_launch(void* const* d_in, const int* in_sizes, int n_in,
                              void* d_out, int out_size, void* d_ws, size_t ws_size,
                              hipStream_t stream)
{
    const float* x   = (const float*)d_in[0];
    const int*   ei  = (const int*)d_in[1];
    const float* W1  = (const float*)d_in[2];
    const float* b1  = (const float*)d_in[3];
    const float* W2  = (const float*)d_in[4];
    const float* b2  = (const float*)d_in[5];
    const float* g1  = (const float*)d_in[6];
    const float* be1 = (const float*)d_in[7];
    const float* m1  = (const float*)d_in[8];
    const float* v1  = (const float*)d_in[9];
    const float* g2  = (const float*)d_in[10];
    const float* be2 = (const float*)d_in[11];
    const float* m2  = (const float*)d_in[12];
    const float* v2  = (const float*)d_in[13];
    const float* Wfc = (const float*)d_in[14];
    const float* bfc = (const float*)d_in[15];
    float* out = (float*)d_out;

    const int N = in_sizes[0] / 128;
    const int E = in_sizes[1] / 2;
    const int* erow = ei;       // edge_index[0]
    const int* ecol = ei + E;   // edge_index[1]

    // Workspace layout (all f32): dinv[N] | buf0[N*128] | buf1[N*128] | buf2[N*128]
    float* dinv = (float*)d_ws;
    float* buf0 = dinv + N;                  // h1, later x1
    float* buf1 = buf0 + (size_t)N * 128;    // agg (both layers)
    float* buf2 = buf1 + (size_t)N * 128;    // h2, later x2

    const int T = 256;
    dim3 blk(T);
    int gN   = (N + T - 1) / T;
    int gNF  = (N * 128 + T - 1) / T;
    int gE   = (E + T - 1) / T;
    int gEw  = (E + 7) / 8;          // 8 edges (waves) per block
    int gTil = (N + 127) / 128;

    size_t shGemm = (size_t)2 * 128 * 136 * sizeof(unsigned short);          // ~70 KB
    size_t shFc   = ((size_t)128 * 392 + 48 * 392) * sizeof(unsigned short); // ~138 KB

    // degrees -> dinv = rsqrt(deg + 1)
    zero_f32<<<gN, blk, 0, stream>>>(dinv, N);
    deg_count<<<gE, blk, 0, stream>>>(erow, dinv, E);
    deg_finish<<<gN, blk, 0, stream>>>(dinv, N);

    // ---- layer 1 ----
    gemm_nodes_bf16<<<gTil, blk, shGemm, stream>>>(x, W1, buf0, N);           // h1
    init_agg<<<gNF, blk, 0, stream>>>(buf0, dinv, b1, buf1, N);               // self-loop + b1
    scatter_edges<<<gEw, blk, 0, stream>>>(erow, ecol, dinv, buf0, buf1, E);  // messages
    bn_relu<<<gNF, blk, 0, stream>>>(buf1, g1, be1, m1, v1, buf0, N);         // x1 -> buf0

    // ---- layer 2 ----
    gemm_nodes_bf16<<<gTil, blk, shGemm, stream>>>(buf0, W2, buf2, N);        // h2
    init_agg<<<gNF, blk, 0, stream>>>(buf2, dinv, b2, buf1, N);
    scatter_edges<<<gEw, blk, 0, stream>>>(erow, ecol, dinv, buf2, buf1, E);
    bn_relu<<<gNF, blk, 0, stream>>>(buf1, g2, be2, m2, v2, buf2, N);         // x2 -> buf2

    // ---- final FC on concat(x, x1, x2) ----
    fc_concat_bf16<<<gTil, blk, shFc, stream>>>(x, buf0, buf2, Wfc, bfc, out, N);
}
